// GeometryAwareSemanticAttention_26173530701965
// MI455X (gfx1250) — matrix operand order repaired and checked by reading, not compiled, not run on hardware
//
#include <hip/hip_runtime.h>

typedef __bf16 bf16;
typedef __attribute__((ext_vector_type(16))) __bf16 v16bf;
typedef __attribute__((ext_vector_type(8)))  float  v8f;

#define D_MODEL 256
#define L_TOK   2048
#define SCALE_F 0.17677669529663687f   /* 1/sqrt(32) */
#define LN_EPS  1e-5f

// ---------------------------------------------------------------------------
// D = A(16x32 bf16) * B(32x16 bf16) + C(16x16 f32)
// emits v_wmma_f32_16x16x32_bf16
// ---------------------------------------------------------------------------
static __device__ __forceinline__ v8f wmma_bf16(v16bf a, v16bf b, v8f c) {
  return __builtin_amdgcn_wmma_f32_16x16x32_bf16(
      /*neg_a=*/false, a, /*neg_b=*/false, b,
      /*c_mod=*/(short)0, c, /*reuse_a=*/false, /*reuse_b=*/false);
}

// Tile layouts used by this code (wave32; contiguous k per lane half, not the ISA table order):
//  A 16x32 bf16 tile, flat index = lane*16 + e:
//      M = lane&15, half = lane>>4, K = e + 16*half
//  B 32x16 bf16 tile, flat index = lane*16 + e:
//      N = lane&15, K = e + 16*(lane>>4)
//  C/D 16x16 f32: element j of v8f -> M = j + 8*(lane>>4), N = lane&15

// ---------------------------------------------------------------------------
// Pack x [2048,256] f32 (row major) into bf16 A-tiles: tile (mt, kt) of 16x32
// ---------------------------------------------------------------------------
__global__ void pack_x_k(const float* __restrict__ x, bf16* __restrict__ xA) {
  int f = blockIdx.x * blockDim.x + threadIdx.x;   // < 2048*256
  int tile = f >> 9, r = f & 511, lane = r >> 4, e = r & 15;
  int mt = tile >> 3, kt = tile & 7;
  int M = lane & 15, half = lane >> 4;
  int K = e + 16 * half;
  xA[f] = (bf16)x[(mt * 16 + M) * D_MODEL + kt * 32 + K];
}

// ---------------------------------------------------------------------------
// Pack W [256,256] f32 (used as x @ W^T, so B(k,n) = W[n][k]) into B-tiles.
// ---------------------------------------------------------------------------
__global__ void pack_w_k(const float* __restrict__ W, bf16* __restrict__ wB) {
  int f = blockIdx.x * blockDim.x + threadIdx.x;   // < 256*256
  int tile = f >> 9, r = f & 511, lane = r >> 4, e = r & 15;
  int nt = tile >> 3, kt = tile & 7;
  int nn = lane & 15;
  int kk = e + 16 * (lane >> 4);
  wB[f] = (bf16)W[(nt * 16 + nn) * D_MODEL + kt * 32 + kk];
}

// ---------------------------------------------------------------------------
// QKV projection: Y = x @ W^T + b.  grid=(128 row-tiles, 3 matrices),
// block=512 (16 waves, wave w owns output column tile nt=w).
// Scatters Q as A-tiles [h][qt], K^T as B-tiles [h][ktile], V as B-tiles
// [h][token-chunk32][d-half] so attention consumes them directly.
// ---------------------------------------------------------------------------
__global__ void qkv_k(const bf16* __restrict__ xA,
                      const bf16* __restrict__ wq, const bf16* __restrict__ wk,
                      const bf16* __restrict__ wv,
                      const float* __restrict__ bq, const float* __restrict__ bk,
                      const float* __restrict__ bv,
                      bf16* __restrict__ qA, bf16* __restrict__ kB,
                      bf16* __restrict__ vB) {
  int mt = blockIdx.x, mat = blockIdx.y;
  int wave = threadIdx.x >> 5, lane = threadIdx.x & 31;
  int nt = wave;
  const bf16*  w    = (mat == 0) ? wq : (mat == 1) ? wk : wv;
  const float* bias = (mat == 0) ? bq : (mat == 1) ? bk : bv;

  v8f acc = {};
#pragma unroll
  for (int kt = 0; kt < 8; ++kt) {
    v16bf a = *(const v16bf*)(xA + (mt * 8 + kt) * 512 + lane * 16);
    v16bf b = *(const v16bf*)(w  + (nt * 8 + kt) * 512 + lane * 16);
    acc = wmma_bf16(a, b, acc);
  }
  union { v8f v; float f[8]; } au; au.v = acc;

  int half = lane >> 4, nloc = lane & 15;
  int n = nt * 16 + nloc;
  float bn = bias[n];
  int h = n >> 5, d = n & 31;
#pragma unroll
  for (int j = 0; j < 8; ++j) {
    int m = j + 8 * half;                  // local row in tile
    bf16 val = (bf16)(au.f[j] + bn);
    if (mat == 0) {                        // Q -> A-tile (mm=m, kk=d)
      int eh = d >> 4;
      int ea = d & 15;
      int la = m + 16 * eh;
      qA[(h * 128 + mt) * 512 + la * 16 + ea] = val;
    } else if (mat == 1) {                 // K^T -> B-tile (kk=d, nn=m)
      int lb = m + 16 * (d >> 4);
      int eb = d & 15;
      kB[(h * 128 + mt) * 512 + lb * 16 + eb] = val;
    } else {                               // V -> B-tile (kk=token, nn=d%16)
      int T = mt * 16 + m;
      int tc = T >> 5, tk = T & 31;
      int dh = d >> 4, nn = d & 15;
      int lb = nn + 16 * (tk >> 4);
      int eb = tk & 15;
      vB[((h * 64 + tc) * 2 + dh) * 512 + lb * 16 + eb] = val;
    }
  }
}

// ---------------------------------------------------------------------------
// Fused attention: one workgroup per (query-tile qt, head h); 4 waves.
// LDS: full 16 x 2048 f32 score stripe (CDNA5: 320KB/WGP allows it).
// ---------------------------------------------------------------------------
__global__ void attn_k(const bf16* __restrict__ qA, const bf16* __restrict__ kB,
                       const bf16* __restrict__ vB,
                       const float* __restrict__ pts, const float* __restrict__ betap,
                       const float* __restrict__ kw1, const float* __restrict__ kb1,
                       const float* __restrict__ kw2, const float* __restrict__ kb2,
                       bf16* __restrict__ attnA) {
  extern __shared__ float sm[];
  float* S     = sm;                   // 16*2048
  float* qp    = S + 16 * 2048;        // 48
  float* w1    = qp + 48;              // 32
  float* b1    = w1 + 32;              // 32
  float* w2    = b1 + 32;              // 32
  float* redA  = w2 + 32;              // 128
  float* redB  = redA + 128;           // 128
  float* rinv  = redB + 128;           // 16
  float* Opart = rinv + 16;            // 4*512

  int qt = blockIdx.x, h = blockIdx.y;
  int tid = threadIdx.x;
  int wave = tid >> 5, lane = tid & 31;
  int half = lane >> 4, nloc = lane & 15;

  if (tid < 32) { w1[tid] = kw1[tid]; b1[tid] = kb1[tid]; w2[tid] = kw2[tid]; }
  if (tid < 48) { qp[tid] = pts[qt * 16 * 3 + tid]; }   // qp[row*3+c]
  __syncthreads();
  float beta  = betap[0];
  float bias2 = kb2[0];

  // ---- Phase 1: scores = (Q K^T)*scale + beta*bias(dist MLP), into LDS ----
  v16bf qa = *(const v16bf*)(qA + (h * 128 + qt) * 512 + lane * 16);
#pragma unroll 1
  for (int i = 0; i < 32; ++i) {
    int kt = wave + 4 * i;
    __builtin_prefetch(kB + (h * 128 + ((kt + 4) & 127)) * 512 + lane * 16, 0, 1);
    v16bf kb = *(const v16bf*)(kB + (h * 128 + kt) * 512 + lane * 16);
    v8f z = {};
    v8f s = wmma_bf16(qa, kb, z);
    union { v8f v; float f[8]; } su; su.v = s;
    int t = kt * 16 + nloc;
    float kx = pts[t * 3 + 0], ky = pts[t * 3 + 1], kz = pts[t * 3 + 2];
#pragma unroll 1
    for (int j = 0; j < 8; ++j) {
      int m = j + 8 * half;
      float dx = qp[m * 3 + 0] - kx, dy = qp[m * 3 + 1] - ky, dz = qp[m * 3 + 2] - kz;
      float sq = dx * dx + dy * dy + dz * dz;
      float dist = sq > 0.f ? sqrtf(sq) : 0.f;
      float bs = bias2;
#pragma unroll
      for (int ii = 0; ii < 32; ++ii) {
        float hh = fmaxf(dist * w1[ii] + b1[ii], 0.f);
        bs += hh * w2[ii];
      }
      bs = fminf(fmaxf(bs, -10.f), 0.f);
      S[m * 2048 + t] = su.f[j] * SCALE_F + beta * bs;
    }
  }
  __syncthreads();

  // ---- Phase 2: row softmax over 2048 (8 threads per row) ----
  int row = tid >> 3, sub = tid & 7;
  int c0 = sub * 256;
  float mx = -3.4e38f;
  for (int c = 0; c < 256; ++c) mx = fmaxf(mx, S[row * 2048 + c0 + c]);
  redA[row * 8 + sub] = mx;
  __syncthreads();
  float rmx = redA[row * 8];
  for (int k = 1; k < 8; ++k) rmx = fmaxf(rmx, redA[row * 8 + k]);
  float sum = 0.f;
  for (int c = 0; c < 256; ++c) {
    float p = __expf(S[row * 2048 + c0 + c] - rmx);
    S[row * 2048 + c0 + c] = p;
    sum += p;
  }
  redB[row * 8 + sub] = sum;
  __syncthreads();
  float rs = 0.f;
  for (int k = 0; k < 8; ++k) rs += redB[row * 8 + k];
  if (sub == 0) rinv[row] = 1.f / rs;
  __syncthreads();

  // ---- Phase 3: O = P @ V  (each wave: 16 chunks of K=32 tokens) ----
  v8f acc0 = {}, acc1 = {};
  int mm = lane & 15;
  float inv = rinv[mm];
#pragma unroll 1
  for (int i = 0; i < 16; ++i) {
    int c = wave + 4 * i;                       // 32-token chunk, 0..63
    union { v16bf v; bf16 s[16]; } pu;
#pragma unroll
    for (int e = 0; e < 16; ++e) {
      int K = e + 16 * half;
      pu.s[e] = (bf16)(S[mm * 2048 + c * 32 + K] * inv);
    }
    v16bf vb0 = *(const v16bf*)(vB + ((h * 64 + c) * 2 + 0) * 512 + lane * 16);
    v16bf vb1 = *(const v16bf*)(vB + ((h * 64 + c) * 2 + 1) * 512 + lane * 16);
    acc0 = wmma_bf16(pu.v, vb0, acc0);
    acc1 = wmma_bf16(pu.v, vb1, acc1);
  }
  union { v8f v; float f[8]; } a0, a1; a0.v = acc0; a1.v = acc1;
  float* Ow = Opart + wave * 512;
#pragma unroll
  for (int j = 0; j < 8; ++j) {
    int m = j + 8 * half;
    Ow[m * 32 + nloc]      = a0.f[j];
    Ow[m * 32 + 16 + nloc] = a1.f[j];
  }
  __syncthreads();

  // cross-wave reduce + scatter O (16x32) as a bf16 A-tile (mt=qt, kt=h)
#pragma unroll
  for (int r = 0; r < 4; ++r) {
    int f = tid * 4 + r;                 // 0..511
    int la = f >> 4, ea = f & 15;
    int M = la & 15, hf = la >> 4;
    int K = ea + 16 * hf;
    float v = Opart[0 * 512 + M * 32 + K] + Opart[1 * 512 + M * 32 + K] +
              Opart[2 * 512 + M * 32 + K] + Opart[3 * 512 + M * 32 + K];
    attnA[(qt * 8 + h) * 512 + f] = (bf16)v;
  }
}

// ---------------------------------------------------------------------------
// Out-projection + residual + LayerNorm. grid=128 row-tiles, block=256.
// Wave w computes column tiles {w, w+8}; LN fused through LDS.
// ---------------------------------------------------------------------------
__global__ void proj_ln_k(const bf16* __restrict__ attnA, const bf16* __restrict__ woB,
                          const float* __restrict__ bo, const float* __restrict__ x,
                          const float* __restrict__ g, const float* __restrict__ b,
                          float* __restrict__ out) {
  __shared__ float Y[16 * 256];
  __shared__ float redm[256], redv[256];
  int mt = blockIdx.x;
  int tid = threadIdx.x;
  int wave = tid >> 5, lane = tid & 31;
  int half = lane >> 4, nloc = lane & 15;

#pragma unroll
  for (int p = 0; p < 2; ++p) {
    int nt = wave + 8 * p;
    v8f acc = {};
#pragma unroll
    for (int kt = 0; kt < 8; ++kt) {
      v16bf a  = *(const v16bf*)(attnA + (mt * 8 + kt) * 512 + lane * 16);
      v16bf bb = *(const v16bf*)(woB   + (nt * 8 + kt) * 512 + lane * 16);
      acc = wmma_bf16(a, bb, acc);
    }
    union { v8f v; float f[8]; } au; au.v = acc;
    int n = nt * 16 + nloc;
    float bn = bo[n];
#pragma unroll
    for (int j = 0; j < 8; ++j) {
      int m = j + 8 * half;
      Y[m * 256 + n] = au.f[j] + bn + x[(mt * 16 + m) * D_MODEL + n];
    }
  }
  __syncthreads();

  int row = tid >> 4, grp = tid & 15;
  float s = 0.f, ss = 0.f;
  for (int c = 0; c < 16; ++c) {
    float y = Y[row * 256 + grp * 16 + c];
    s += y; ss += y * y;
  }
  redm[row * 16 + grp] = s;
  redv[row * 16 + grp] = ss;
  __syncthreads();
  float mean = 0.f, m2 = 0.f;
  for (int k = 0; k < 16; ++k) { mean += redm[row * 16 + k]; m2 += redv[row * 16 + k]; }
  mean *= (1.f / 256.f);
  float var = m2 * (1.f / 256.f) - mean * mean;
  float rstd = rsqrtf(var + LN_EPS);
  for (int c = 0; c < 16; ++c) {
    int col = grp * 16 + c;
    float y = Y[row * 256 + col];
    out[(mt * 16 + row) * D_MODEL + col] = (y - mean) * rstd * g[col] + b[col];
  }
}

// ---------------------------------------------------------------------------
extern "C" void kernel_launch(void* const* d_in, const int* in_sizes, int n_in,
                              void* d_out, int out_size, void* d_ws, size_t ws_size,
                              hipStream_t stream) {
  (void)in_sizes; (void)n_in; (void)out_size; (void)ws_size;
  const float* x    = (const float*)d_in[0];
  const float* pts  = (const float*)d_in[1];
  const float* Wq   = (const float*)d_in[2];
  const float* bq   = (const float*)d_in[3];
  const float* Wk   = (const float*)d_in[4];
  const float* bk   = (const float*)d_in[5];
  const float* Wv   = (const float*)d_in[6];
  const float* bv   = (const float*)d_in[7];
  const float* Wo   = (const float*)d_in[8];
  const float* bo   = (const float*)d_in[9];
  const float* beta = (const float*)d_in[10];
  const float* kw1  = (const float*)d_in[11];
  const float* kb1  = (const float*)d_in[12];
  const float* kw2  = (const float*)d_in[13];
  const float* kb2  = (const float*)d_in[14];
  const float* lng  = (const float*)d_in[15];
  const float* lnb  = (const float*)d_in[16];
  float* out = (float*)d_out;

  bf16* p     = (bf16*)d_ws;
  bf16* xA    = p; p += 2048 * 256;       // x, A-tiles
  bf16* wqB   = p; p += 256 * 256;        // weights, B-tiles
  bf16* wkB   = p; p += 256 * 256;
  bf16* wvB   = p; p += 256 * 256;
  bf16* woB   = p; p += 256 * 256;
  bf16* qA    = p; p += 8 * 128 * 512;    // Q A-tiles per head
  bf16* kBuf  = p; p += 8 * 128 * 512;    // K^T B-tiles per head
  bf16* vBuf  = p; p += 8 * 64 * 2 * 512; // V B-tiles per head
  bf16* attnA = p; p += 2048 * 256;       // attention output, A-tiles

  pack_x_k<<<2048, 256, 0, stream>>>(x, xA);
  pack_w_k<<<256, 256, 0, stream>>>(Wq, wqB);
  pack_w_k<<<256, 256, 0, stream>>>(Wk, wkB);
  pack_w_k<<<256, 256, 0, stream>>>(Wv, wvB);
  pack_w_k<<<256, 256, 0, stream>>>(Wo, woB);
  qkv_k<<<dim3(128, 3), 512, 0, stream>>>(xA, wqB, wkB, wvB, bq, bk, bv,
                                          qA, kBuf, vBuf);
  size_t smem = (size_t)(16 * 2048 + 48 + 3 * 32 + 128 + 128 + 16 + 4 * 512) *
                sizeof(float);
  attn_k<<<dim3(128, 8), 128, smem, stream>>>(qA, kBuf, vBuf, pts, beta,
                                              kw1, kb1, kw2, kb2, attnA);
  proj_ln_k<<<128, 256, 0, stream>>>(attnA, woB, bo, x, lng, lnb, out);
}
